// Encoder_66494683676993
// MI455X (gfx1250) — compile-verified, matrix operand order checked by reading, not run on hardware
//
#include <hip/hip_runtime.h>
#include <hip/hip_bf16.h>

typedef __attribute__((ext_vector_type(16))) _Float16 v16h;
typedef __attribute__((ext_vector_type(8)))  float    v8f;
typedef _Float16 half_t;

// ---------------------------------------------------------------------------
// Weight convert + transpose: W[27][Cin][128] f32  ->  Wt[27][128][Cin] f16
// ---------------------------------------------------------------------------
__global__ void wconvt_kernel(const float* __restrict__ W, half_t* __restrict__ Wt,
                              int Cin, int total) {
    int i = blockIdx.x * blockDim.x + threadIdx.x;
    if (i >= total) return;
    int co = i % 128;
    int ci = (i / 128) % Cin;
    int k  = i / (128 * Cin);
    Wt[((size_t)k * 128 + co) * Cin + ci] = (half_t)W[i];
}

// ---------------------------------------------------------------------------
// conv0: x[N0,3] f32, km[27,N0], W[27,3,64], b[64] -> out[N0,64] f16
// Tiny K (=3): scalar VALU is the right tool; WMMA K=32 would waste 10x.
// ---------------------------------------------------------------------------
__global__ void conv0_kernel(const float* __restrict__ x, const int* __restrict__ km,
                             const float* __restrict__ W, const float* __restrict__ b,
                             half_t* __restrict__ out, int N) {
    int t = blockIdx.x * blockDim.x + threadIdx.x;
    if (t >= N * 64) return;
    int n = t >> 6, c = t & 63;
    float acc = b[c];
    #pragma unroll 1
    for (int k = 0; k < 27; ++k) {
        int idx = km[k * N + n];
        if (idx >= 0) {
            const float* f = x + (size_t)idx * 3;
            const float* w = W + k * 192 + c;   // [27][3][64]
            acc = fmaf(f[0], w[0],   acc);
            acc = fmaf(f[1], w[64],  acc);
            acc = fmaf(f[2], w[128], acc);
        }
    }
    out[t] = (half_t)acc;
}

// ---------------------------------------------------------------------------
// WMMA sparse conv: feat[Nin,CIN] f16, kmap[27,Nout], Wt[27,128,CIN] f16,
// bias[128] f32 -> out_f32[Nout,128] (opt) and/or out_f16[Nout,128] (opt).
//
// Block = 32 output points x 128 out-channels, 256 threads = 8 waves.
// Wave w owns out-channel slice [16w,16w+16) and both 16-row sub-tiles
// (acc0: rows 0..15, acc1: rows 16..31) so each B fragment is reused twice.
// LDS A-tile is double-buffered: per tap, gather tap k+1 into registers,
// run tap k's WMMAs from LDS, ds_store the prefetched rows, ONE barrier.
// ---------------------------------------------------------------------------
template <int CIN>
__global__ __launch_bounds__(256) void sconv_wmma_kernel(
    const half_t* __restrict__ feat,
    const int*    __restrict__ kmap,
    const half_t* __restrict__ Wt,
    const float*  __restrict__ bias,
    float*        __restrict__ out_f32,
    half_t*       __restrict__ out_f16,
    int Nout, int relu)
{
    constexpr int STRIDE = CIN + 8;            // halves; conflict-free row stride
    constexpr int BUFH   = 32 * STRIDE;        // halves per LDS buffer
    constexpr int CH     = CIN / 8;            // halves staged per thread (16 or 8)
    __shared__ half_t lds[2 * BUFH];

    const int tid  = threadIdx.x;
    const int lane = tid & 31;
    const int wave = tid >> 5;                 // 0..7
    const int base = blockIdx.x * 32;          // first output point of tile

    const int c   = lane & 15;                 // column within 16-wide slice
    const int hi  = lane >> 4;                 // lane group (K phase / row phase)
    const int col = wave * 16 + c;             // output channel 0..127

    const int p     = tid >> 3;                // staged point 0..31
    const int chunk = tid & 7;                 // chunk within row

    v8f acc0 = {}, acc1 = {};                  // f32 accumulators (bias at end)

    uint4 r0, r1;                              // gather prefetch registers
    auto fetch = [&](int k) {
        int n   = base + p;
        int idx = (n < Nout) ? kmap[k * Nout + n] : -1;
        r0.x = r0.y = r0.z = r0.w = 0u; r1 = r0;
        if (idx >= 0) {
            const uint4* src = (const uint4*)(feat + (size_t)idx * CIN + chunk * CH);
            r0 = src[0];
            if (CH == 16) r1 = src[1];
        }
    };
    auto stage = [&](int k1) {
        half_t* dst = &lds[(k1 & 1) * BUFH + p * STRIDE + chunk * CH];
        *(uint4*)dst = r0;
        if (CH == 16) *(uint4*)(dst + 8) = r1;
    };

    fetch(0); stage(0);
    __syncthreads();

    for (int k = 0; k < 27; ++k) {
        const half_t* cur = &lds[(k & 1) * BUFH];
        const bool more = (k + 1 < 27);
        if (more) fetch(k + 1);                // global gathers overlap WMMAs below

        const half_t* wrow = Wt + ((size_t)k * 128 + col) * CIN;
        #pragma unroll
        for (int kk = 0; kk < CIN / 32; ++kk) {
            v16h a0, a1, b;
            // B fragment: lane holds column `col`; Wt row contiguous in K
            const half_t* brow = wrow + kk * 32 + hi * 8;
            *(uint4*)&b         = *(const uint4*)(brow);
            *(((uint4*)&b) + 1) = *(const uint4*)(brow + 16);
            // A fragments: lane holds row M=c (sub-tile 0) and M=16+c (sub-tile 1)
            const half_t* ar0 = &cur[c * STRIDE + kk * 32 + hi * 8];
            *(uint4*)&a0         = *(const uint4*)(ar0);
            *(((uint4*)&a0) + 1) = *(const uint4*)(ar0 + 16);
            const half_t* ar1 = ar0 + 16 * STRIDE;
            *(uint4*)&a1         = *(const uint4*)(ar1);
            *(((uint4*)&a1) + 1) = *(const uint4*)(ar1 + 16);
            acc0 = __builtin_amdgcn_wmma_f32_16x16x32_f16(
                       false, a0, false, b, (short)0, acc0, false, false);
            acc1 = __builtin_amdgcn_wmma_f32_16x16x32_f16(
                       false, a1, false, b, (short)0, acc1, false, false);
        }
        if (more) stage(k + 1);                // write OTHER buffer; no extra barrier
        __syncthreads();                       // single barrier per tap
    }

    // Epilogue: D layout -> lane holds (M = r + 8*hi, N = col) in acc[r]
    const float bv = bias[col];
    #pragma unroll
    for (int s = 0; s < 2; ++s) {
        v8f acc = s ? acc1 : acc0;
        #pragma unroll
        for (int r = 0; r < 8; ++r) {
            int n = base + s * 16 + r + hi * 8;
            if (n < Nout) {
                float v = acc[r] + bv;
                if (relu) v = fmaxf(v, 0.0f);
                size_t off = (size_t)n * 128 + col;
                if (out_f32) out_f32[off] = v;
                if (out_f16) out_f16[off] = (half_t)v;
            }
        }
    }
}

// ---------------------------------------------------------------------------
static inline unsigned cdiv(long long a, long long b) { return (unsigned)((a + b - 1) / b); }

extern "C" void kernel_launch(void* const* d_in, const int* in_sizes, int n_in,
                              void* d_out, int out_size, void* d_ws, size_t ws_size,
                              hipStream_t stream) {
    const float* x    = (const float*)d_in[0];
    const int*   km0  = (const int*)d_in[1];
    const int*   kmd0 = (const int*)d_in[2];
    const int*   km1  = (const int*)d_in[3];
    const int*   kmd1 = (const int*)d_in[4];
    const int*   km2  = (const int*)d_in[5];
    const int*   kmd2 = (const int*)d_in[6];
    const float* W0  = (const float*)d_in[7];  const float* b0  = (const float*)d_in[8];
    const float* Wd0 = (const float*)d_in[9];  const float* bd0 = (const float*)d_in[10];
    const float* W1  = (const float*)d_in[11]; const float* b1  = (const float*)d_in[12];
    const float* Wd1 = (const float*)d_in[13]; const float* bd1 = (const float*)d_in[14];
    const float* W2  = (const float*)d_in[15]; const float* b2  = (const float*)d_in[16];
    const float* Wd2 = (const float*)d_in[17]; const float* bd2 = (const float*)d_in[18];

    const int N0 = in_sizes[0] / 3;
    const int N1 = in_sizes[2] / 27;
    const int N2 = in_sizes[4] / 27;
    const int N3 = in_sizes[6] / 27;

    // d_out = [out2 (N3x128), out1 (N2x128), out0 (N1x128)] f32
    float* out2 = (float*)d_out;
    float* out1 = out2 + (size_t)N3 * 128;
    float* out0 = out1 + (size_t)N2 * 128;

    // Workspace carve-up (halves, all offsets 128B-aligned)
    half_t* ws = (half_t*)d_ws;
    size_t o = 0;
    half_t* wt_d0 = ws + o; o += (size_t)27 * 128 * 64;
    half_t* wt_1  = ws + o; o += (size_t)27 * 128 * 128;
    half_t* wt_d1 = ws + o; o += (size_t)27 * 128 * 128;
    half_t* wt_2  = ws + o; o += (size_t)27 * 128 * 128;
    half_t* wt_d2 = ws + o; o += (size_t)27 * 128 * 128;
    size_t S = (size_t)N0 * 64;
    if ((size_t)N1 * 128 > S) S = (size_t)N1 * 128;
    if ((size_t)N2 * 128 > S) S = (size_t)N2 * 128;
    S = (S + 63) & ~(size_t)63;
    half_t* bufA = ws + o; o += S;
    half_t* bufB = ws + o;

    // Convert + transpose weights to f16 [27][128][Cin]
    wconvt_kernel<<<cdiv(27LL * 64 * 128, 256), 256, 0, stream>>>(Wd0, wt_d0, 64, 27 * 64 * 128);
    wconvt_kernel<<<cdiv(27LL * 128 * 128, 256), 256, 0, stream>>>(W1,  wt_1,  128, 27 * 128 * 128);
    wconvt_kernel<<<cdiv(27LL * 128 * 128, 256), 256, 0, stream>>>(Wd1, wt_d1, 128, 27 * 128 * 128);
    wconvt_kernel<<<cdiv(27LL * 128 * 128, 256), 256, 0, stream>>>(W2,  wt_2,  128, 27 * 128 * 128);
    wconvt_kernel<<<cdiv(27LL * 128 * 128, 256), 256, 0, stream>>>(Wd2, wt_d2, 128, 27 * 128 * 128);

    // conv0: x -> h0 (f16) in bufA
    conv0_kernel<<<cdiv((long long)N0 * 64, 256), 256, 0, stream>>>(x, km0, W0, b0, bufA, N0);

    // down0: h0 -> out0 (f32, relu) + f16 copy in bufB
    sconv_wmma_kernel<64><<<cdiv(N1, 32), 256, 0, stream>>>(
        bufA, kmd0, wt_d0, bd0, out0, bufB, N1, 1);

    // conv1: out0 -> h1 (f16 only) in bufA
    sconv_wmma_kernel<128><<<cdiv(N1, 32), 256, 0, stream>>>(
        bufB, km1, wt_1, b1, (float*)nullptr, bufA, N1, 0);

    // down1: h1 -> out1 (f32, relu) + f16 copy in bufB
    sconv_wmma_kernel<128><<<cdiv(N2, 32), 256, 0, stream>>>(
        bufA, kmd1, wt_d1, bd1, out1, bufB, N2, 1);

    // conv2: out1 -> h2 (f16 only) in bufA
    sconv_wmma_kernel<128><<<cdiv(N2, 32), 256, 0, stream>>>(
        bufB, km2, wt_2, b2, (float*)nullptr, bufA, N2, 0);

    // down2: h2 -> out2 (f32, no relu)
    sconv_wmma_kernel<128><<<cdiv(N3, 32), 256, 0, stream>>>(
        bufA, kmd2, wt_d2, bd2, out2, (half_t*)nullptr, N3, 0);
}
